// DotProdSelfAttention_66907000537404
// MI455X (gfx1250) — compile-verified, matrix operand order checked by reading, not run on hardware
//
#include <hip/hip_runtime.h>

// ---------------------------------------------------------------------------
// DotProdSelfAttention on MI455X (gfx1250, wave32, WMMA)
//   B=8, S=2048, D=512, UNITS=512, fp32 in/out.
// Strategy:
//   Kernel 1: QKV projection GEMM (bf16 WMMA, f32 acc). Emits bf16 workspace:
//             Q (scaled by 1/sqrt(512)) and K row-major, V transposed [b][u][s].
//   Kernel 2: per (batch, 16-query-row) block: TDM-staged Q tile, full-row
//             scores in ~210KB LDS, exact softmax, then P x V with per-wave
//             64-column output slices. Q fragments live in registers.
// ---------------------------------------------------------------------------

typedef __bf16 bf16;
typedef __attribute__((ext_vector_type(2)))  __bf16 v2bf;
typedef __attribute__((ext_vector_type(8)))  __bf16 v8bf;
typedef __attribute__((ext_vector_type(16))) __bf16 v16bf;
typedef __attribute__((ext_vector_type(8)))  float  v8f;
typedef __attribute__((ext_vector_type(4)))  float  v4f;
typedef __attribute__((ext_vector_type(4)))  unsigned int u32x4;
typedef __attribute__((ext_vector_type(8)))  int   i32x8;
typedef __attribute__((ext_vector_type(4)))  int   i32x4;

#define NB 8
#define NS 2048
#define ND 512
#define NU 512

__device__ __forceinline__ v16bf cat8(v8bf lo, v8bf hi) {
  return __builtin_shufflevector(lo, hi, 0,1,2,3,4,5,6,7,8,9,10,11,12,13,14,15);
}

// Load a 16x32 bf16 WMMA operand (A rows / B columns) from a row-pitched
// buffer. Lane 0-15: row=lane, K in {0..7, 16..23}; lane 16-31: row=lane-16,
// K in {8..15, 24..31}. Two 16-byte loads per lane.
__device__ __forceinline__ v16bf load_frag(const bf16* base, int stride, int lane) {
  int r = lane & 15;
  const bf16* p = base + r * stride + ((lane >> 4) ? 8 : 0);
  v8bf lo = *(const v8bf*)p;
  v8bf hi = *(const v8bf*)(p + 16);
  return cat8(lo, hi);
}

__device__ __forceinline__ v8f wmma_bf16(v16bf a, v16bf b, v8f c) {
  return __builtin_amdgcn_wmma_f32_16x16x32_bf16(
      /*neg_a=*/false, a, /*neg_b=*/false, b,
      /*c_mod=*/(short)0, c, /*reuse_a=*/false, /*reuse_b=*/false);
}

// ---------------------------------------------------------------------------
// Kernel 1: fused QKV projection. grid=(NU/64, (NB*NS)/128, 3), block=256.
//   z=0 -> Q = (x Wq) * 1/sqrt(512), row-major bf16
//   z=1 -> K =  x Wk,               row-major bf16
//   z=2 -> Vt = (x Wv)^T            [b][u][s]  bf16
// ---------------------------------------------------------------------------
__global__ void __launch_bounds__(256)
qkv_proj_kernel(const float* __restrict__ x,
                const float* __restrict__ Wq,
                const float* __restrict__ Wk,
                const float* __restrict__ Wv,
                bf16* __restrict__ Qs,
                bf16* __restrict__ Ks,
                bf16* __restrict__ Vt) {
  __shared__ bf16 As[128][40];   // 128 x 32 used, pitch padded (bank spread)
  __shared__ bf16 Bt[64][40];    // [n][k], 64 x 32 used

  const int z  = blockIdx.z;
  const float* W = (z == 0) ? Wq : (z == 1) ? Wk : Wv;
  const int n0 = blockIdx.x * 64;
  const int m0 = blockIdx.y * 128;
  const int tid  = threadIdx.x;
  const int wave = tid >> 5;
  const int lane = tid & 31;
  const int wm = (wave & 3) * 32;   // M sub-tile base inside block tile
  const int wn = (wave >> 2) * 32;  // N sub-tile base inside block tile

  v8f acc[2][2];
  #pragma unroll
  for (int i = 0; i < 2; ++i)
    #pragma unroll
    for (int j = 0; j < 2; ++j)
      acc[i][j] = (v8f){0.f,0.f,0.f,0.f,0.f,0.f,0.f,0.f};

  // staging coordinates (hoisted)
  const int arow = tid >> 1;            // 0..127
  const int acol = (tid & 1) * 16;      // 0 or 16
  const int bn   = tid & 63;            // 0..63
  const int bk   = (tid >> 6) * 8;      // 0,8,16,24

  for (int k0 = 0; k0 < ND; k0 += 32) {
    __syncthreads();
    // stage x tile: 128 x 32 fp32 -> bf16, two 16B LDS stores per thread
    {
      const float* src = x + (size_t)(m0 + arow) * ND + k0 + acol;
      v8bf p0, p1;
      #pragma unroll
      for (int e = 0; e < 8; ++e) { p0[e] = (bf16)src[e]; p1[e] = (bf16)src[8 + e]; }
      *(v8bf*)&As[arow][acol]     = p0;
      *(v8bf*)&As[arow][acol + 8] = p1;
    }
    // stage W tile transposed: thread owns column n, 8 consecutive k.
    // Global reads are lane-coalesced (n contiguous); LDS store is one b128.
    {
      const float* src = W + (size_t)(k0 + bk) * NU + n0 + bn;
      v8bf pk;
      #pragma unroll
      for (int e = 0; e < 8; ++e) pk[e] = (bf16)src[(size_t)e * NU];
      *(v8bf*)&Bt[bn][bk] = pk;
    }
    __syncthreads();

    v16bf a0 = load_frag(&As[wm     ][0], 40, lane);
    v16bf a1 = load_frag(&As[wm + 16][0], 40, lane);
    v16bf b0 = load_frag(&Bt[wn     ][0], 40, lane);
    v16bf b1 = load_frag(&Bt[wn + 16][0], 40, lane);
    acc[0][0] = wmma_bf16(a0, b0, acc[0][0]);
    acc[0][1] = wmma_bf16(a0, b1, acc[0][1]);
    acc[1][0] = wmma_bf16(a1, b0, acc[1][0]);
    acc[1][1] = wmma_bf16(a1, b1, acc[1][1]);
  }

  const int r    = lane & 15;
  const int half = lane >> 4;
  if (z < 2) {
    const float scl = (z == 0) ? 0.044194173824159216f : 1.0f; // 1/sqrt(512)
    bf16* out = (z == 0) ? Qs : Ks;
    #pragma unroll
    for (int mi = 0; mi < 2; ++mi)
      #pragma unroll
      for (int ni = 0; ni < 2; ++ni)
        #pragma unroll
        for (int rr = 0; rr < 8; ++rr) {
          int m = m0 + wm + mi * 16 + rr + 8 * half;
          int n = n0 + wn + ni * 16 + r;
          out[(size_t)m * NU + n] = (bf16)(acc[mi][ni][rr] * scl);
        }
  } else {
    // Vt[b][u][s]: per lane, 8 contiguous s values -> one 16B store
    const int bidx  = m0 / NS;
    const int sbase = m0 % NS;
    #pragma unroll
    for (int mi = 0; mi < 2; ++mi)
      #pragma unroll
      for (int ni = 0; ni < 2; ++ni) {
        int u = n0 + wn + ni * 16 + r;
        int s = sbase + wm + mi * 16 + 8 * half;
        v8bf pk;
        #pragma unroll
        for (int rr = 0; rr < 8; ++rr) pk[rr] = (bf16)acc[mi][ni][rr];
        *(v8bf*)(Vt + ((size_t)bidx * NU + u) * NS + s) = pk;
      }
  }
}

// ---------------------------------------------------------------------------
// Kernel 2: attention. grid=(NS/16, NB), block=256, dynamic LDS ~210KB:
//   QL : bf16[16][520]   query tile (pitch 520 = TDM pad: 256 DW + 4 DW pad)
//   PL : bf16[16][2056]  softmax probabilities (bf16, for PV WMMA)
//   SL : f32 [16][2048]  raw scores / exp values; reused as f32 O staging
//   red: f32 [2][16][16] row reductions
// ---------------------------------------------------------------------------
#define QL_OFF   0
#define PL_OFF   16640                 // 16*520*2
#define SL_OFF   (16640 + 65792)       // + 16*2056*2  = 82432
#define RED_OFF  (82432 + 131072)      // + 16*2048*4  = 213504
#define SMEM_TOTAL (213504 + 2048)     // 215552 bytes

__global__ void __launch_bounds__(256)
attention_kernel(const bf16* __restrict__ Qs,
                 const bf16* __restrict__ Ks,
                 const bf16* __restrict__ Vt,
                 float* __restrict__ out) {
  extern __shared__ char smem[];
  bf16*  QL  = (bf16*)(smem + QL_OFF);
  bf16*  PL  = (bf16*)(smem + PL_OFF);
  float* SL  = (float*)(smem + SL_OFF);
  float* red = (float*)(smem + RED_OFF);

  const int b   = blockIdx.y;
  const int q0  = blockIdx.x * 16;
  const int tid  = threadIdx.x;
  const int wave = tid >> 5;
  const int lane = tid & 31;

  // ---- stage Q tile (16 x 512 bf16) into QL with 520-element pitch -------
#if __has_builtin(__builtin_amdgcn_tensor_load_to_lds) && \
    __has_builtin(__builtin_amdgcn_s_wait_tensorcnt)
  if (wave == 0) {
    // Tensor DMA: 2D tile 512(x) x 16(y) of 2-byte elements, global->LDS,
    // with LDS padding of 4 DWORDs every 256 DWORDs (=> 520-element pitch).
    const unsigned long long gaddr =
        (unsigned long long)(uintptr_t)(Qs + ((size_t)b * NS + q0) * NU);
    const unsigned int lds_addr = (unsigned int)(uintptr_t)QL; // LDS offset
    u32x4 g0;
    g0[0] = 1u;                                   // count=1, user descriptor
    g0[1] = lds_addr;                             // lds_addr
    g0[2] = (unsigned int)(gaddr & 0xFFFFFFFFu);  // global_addr[31:0]
    g0[3] = (unsigned int)((gaddr >> 32) & 0x01FFFFFFu) | (2u << 30); // type=2
    i32x8 g1;
    g1[0] = (int)((1u << 16)        // data_size = 2 bytes
                | (1u << 20)        // pad_enable
                | (7u << 22)        // pad_interval: 256 DWORDs
                | (3u << 25));      // pad_amount: 4 DWORDs
    g1[1] = (int)((unsigned)(NU & 0xFFFF) << 16); // tensor_dim0[15:0] in hi16
    g1[2] = (int)(((unsigned)NS & 0xFFFF) << 16); // dim0 hi=0, tensor_dim1 lo16
    g1[3] = (int)((unsigned)NU << 16);            // dim1 hi=0, tile_dim0=512
    g1[4] = 16;                                   // tile_dim1=16, tile_dim2=0
    g1[5] = NU;                                   // tensor_dim0_stride lo32
    g1[6] = 0;
    g1[7] = 0;
    i32x4 gz4 = {0, 0, 0, 0};
    i32x8 gz8 = {0, 0, 0, 0, 0, 0, 0, 0};
    __builtin_amdgcn_tensor_load_to_lds(g0, g1, gz4, gz4, gz8, 0);
    __builtin_amdgcn_s_wait_tensorcnt(0);
  }
#else
  {
    int row = tid >> 4;
    int cb  = (tid & 15) * 32;
    const v8bf* src = (const v8bf*)(Qs + ((size_t)b * NS + q0 + row) * NU + cb);
    v8bf* dst = (v8bf*)(QL + row * 520 + cb);
    #pragma unroll
    for (int e = 0; e < 4; ++e) dst[e] = src[e];
  }
#endif
  __syncthreads();

  // ---- phase 1: scores S[16][2048] = (Q/sqrt(d)) K^T ---------------------
  // Q fragments depend only on (uc, lane): hoist all 16 into registers
  // (128 VGPRs) so the key loop is pure global b128 loads + WMMA.
  v16bf qf[16];
  #pragma unroll
  for (int uc = 0; uc < 16; ++uc) qf[uc] = load_frag(QL + uc * 32, 520, lane);

  const bf16* Kb = Ks + (size_t)b * NS * NU;
  for (int kt = wave; kt < NS / 16; kt += 8) {
    v8f acc = (v8f){0.f,0.f,0.f,0.f,0.f,0.f,0.f,0.f};
    const int key = kt * 16 + (lane & 15);
    const bf16* kp = Kb + (size_t)key * NU + ((lane >> 4) ? 8 : 0);
    // prefetch this lane's K row for the next key tile handled by this wave
    if (kt + 8 < NS / 16) __builtin_prefetch(kp + (size_t)128 * NU, 0, 1);
    #pragma unroll
    for (int uc = 0; uc < 16; ++uc) {
      v16bf bb = cat8(*(const v8bf*)(kp + uc * 32),
                      *(const v8bf*)(kp + uc * 32 + 16));
      acc = wmma_bf16(qf[uc], bb, acc);
    }
    const int col = kt * 16 + (lane & 15);
    const int rb  = (lane >> 4) * 8;
    #pragma unroll
    for (int rr = 0; rr < 8; ++rr) SL[(rb + rr) * NS + col] = acc[rr];
  }
  __syncthreads();

  // ---- phase 2: exact row softmax (full 2048-wide rows in LDS) -----------
  {
    const int row = tid >> 4;
    const int sub = (tid & 15) * 2;
    float* srow = SL + row * NS;
    float m = -3.402823466e38f;
    for (int j = sub; j < NS; j += 32) m = fmaxf(m, fmaxf(srow[j], srow[j + 1]));
    red[row * 16 + (tid & 15)] = m;
    __syncthreads();
    float rmax = red[row * 16];
    #pragma unroll
    for (int t = 1; t < 16; ++t) rmax = fmaxf(rmax, red[row * 16 + t]);
    float s = 0.f;
    for (int j = sub; j < NS; j += 32) {
      float e0 = __expf(srow[j]     - rmax);
      float e1 = __expf(srow[j + 1] - rmax);
      srow[j] = e0; srow[j + 1] = e1;
      s += e0 + e1;
    }
    red[256 + row * 16 + (tid & 15)] = s;
    __syncthreads();
    float rsum = 0.f;
    #pragma unroll
    for (int t = 0; t < 16; ++t) rsum += red[256 + row * 16 + t];
    const float inv = 1.0f / rsum;
    bf16* prow = PL + row * 2056;
    for (int j = sub; j < NS; j += 32) {
      v2bf p; p[0] = (bf16)(srow[j] * inv); p[1] = (bf16)(srow[j + 1] * inv);
      *(v2bf*)&prow[j] = p;
    }
  }
  __syncthreads();

  // ---- phase 3: O = P V. Each wave owns 64 output columns. ---------------
  const int n0w = wave * 64;
  const bf16* Vb = Vt + (size_t)b * NU * NS;
  v8f oacc[4];
  #pragma unroll
  for (int t = 0; t < 4; ++t) oacc[t] = (v8f){0.f,0.f,0.f,0.f,0.f,0.f,0.f,0.f};

  for (int kk = 0; kk < NS; kk += 32) {
    v16bf a = load_frag(PL + kk, 2056, lane);
    #pragma unroll
    for (int ti = 0; ti < 4; ++ti) {
      int u = n0w + ti * 16 + (lane & 15);
      const bf16* vp = Vb + (size_t)u * NS + kk + ((lane >> 4) ? 8 : 0);
      v16bf bb = cat8(*(const v8bf*)vp, *(const v8bf*)(vp + 16));
      oacc[ti] = wmma_bf16(a, bb, oacc[ti]);
    }
  }

  // ---- epilogue: stage O tile in LDS (SL is free), then coalesced stores -
  {
    const int r  = lane & 15;
    const int rb = (lane >> 4) * 8;
    #pragma unroll
    for (int ti = 0; ti < 4; ++ti)
      #pragma unroll
      for (int rr = 0; rr < 8; ++rr)
        SL[(rb + rr) * 520 + n0w + ti * 16 + r] = oacc[ti][rr];
  }
  __syncthreads();
  {
    const int row = tid >> 4;
    const int cb  = (tid & 15) * 32;
    const float* srcr = SL + row * 520 + cb;
    float* dstr = out + ((size_t)b * NS + q0 + row) * NU + cb;
    #pragma unroll
    for (int e = 0; e < 8; ++e) ((v4f*)dstr)[e] = ((const v4f*)srcr)[e];
  }
}

// ---------------------------------------------------------------------------
extern "C" void kernel_launch(void* const* d_in, const int* in_sizes, int n_in,
                              void* d_out, int out_size, void* d_ws, size_t ws_size,
                              hipStream_t stream) {
  (void)in_sizes; (void)n_in; (void)out_size; (void)ws_size;
  const float* x  = (const float*)d_in[0];
  const float* Wq = (const float*)d_in[1];
  const float* Wk = (const float*)d_in[2];
  const float* Wv = (const float*)d_in[3];
  float* out = (float*)d_out;

  const size_t qkv_elems = (size_t)NB * NS * NU;
  bf16* Qs = (bf16*)d_ws;
  bf16* Ks = Qs + qkv_elems;
  bf16* Vt = Ks + qkv_elems;

  dim3 g1(NU / 64, (NB * NS) / 128, 3);   // (8, 128, 3)
  qkv_proj_kernel<<<g1, 256, 0, stream>>>(x, Wq, Wk, Wv, Qs, Ks, Vt);

  static_assert(SMEM_TOTAL <= 320 * 1024, "fits CDNA5 WGP LDS");
  (void)hipFuncSetAttribute((const void*)attention_kernel,
                            hipFuncAttributeMaxDynamicSharedMemorySize,
                            SMEM_TOTAL);
  dim3 g2(NS / 16, NB);                   // (128, 8)
  attention_kernel<<<g2, 256, SMEM_TOTAL, stream>>>(Qs, Ks, Vt, out);
}
